// GNN_12532714570571
// MI455X (gfx1250) — matrix operand order repaired and checked by reading, not compiled, not run on hardware
//
#include <hip/hip_runtime.h>

#define C_DIM 128
#define NGRAPH 512

typedef __attribute__((ext_vector_type(16))) _Float16 v16h;
typedef __attribute__((ext_vector_type(8)))  float    v8f;

// ---------------------------------------------------------------- degree / norm
__global__ void init_deg_kernel(float* __restrict__ deg, int N) {
    int i = blockIdx.x * blockDim.x + threadIdx.x;
    if (i < N) deg[i] = 1.0f;                      // self-loop weight
}

__global__ void accum_deg_kernel(const int* __restrict__ col,
                                 const float* __restrict__ w,
                                 float* __restrict__ deg, int E) {
    int e = blockIdx.x * blockDim.x + threadIdx.x;
    if (e < E) atomicAdd(&deg[col[e]], w[e]);
}

__global__ void dis_kernel(const float* __restrict__ deg,
                           float* __restrict__ dis, int N) {
    int i = blockIdx.x * blockDim.x + threadIdx.x;
    if (i < N) {
        float d = deg[i];
        dis[i] = (d > 0.0f) ? rsqrtf(d) : 0.0f;
    }
}

// ---------------------------------------------------------------- WMMA GEMM
// dst[N,128] = src[N,128] @ W[128,128], f16 inputs, f32 accumulate.
// One wave computes one 16-row strip (8 tiles of 16x16, K=128 in 4 chunks of 32).
__global__ void __launch_bounds__(256)
gemm_wmma_kernel(const float* __restrict__ src, const float* __restrict__ W,
                 float* __restrict__ dst, int nrows) {
    __shared__ __align__(32) _Float16 Wt[C_DIM * C_DIM];   // Wt[n][k] = W[k][n], 32KB LDS

    // stage + transpose weights (f32 -> f16)
    for (int idx = threadIdx.x; idx < C_DIM * C_DIM; idx += blockDim.x) {
        int k = idx >> 7, n = idx & (C_DIM - 1);
        Wt[n * C_DIM + k] = (_Float16)W[idx];
    }
    __syncthreads();

    const int wave   = threadIdx.x >> 5;
    const int lane   = threadIdx.x & 31;
    const int mtile  = blockIdx.x * 8 + wave;
    const int mtiles = nrows >> 4;                 // nrows divisible by 16 (N=100000)
    if (mtile >= mtiles) return;                   // wave-uniform: EXEC stays all-1s

    const int mrow = lane & 15;
    const int half = lane >> 4;
    const float* arow = src + (size_t)(mtile * 16 + mrow) * C_DIM;

    // A fragments: lane holds row M=mrow; lane-half selects
    // K = kk*32 + half*16 + j (j<16), matching B's contiguous K order.
    v16h afrag[4];
    #pragma unroll
    for (int kk = 0; kk < 4; ++kk) {
        const int k0 = kk * 32 + half * 16;
        const int k1 = kk * 32 + half * 16 + 8;
        #pragma unroll
        for (int j = 0; j < 8; ++j) {
            afrag[kk][j]     = (_Float16)arow[k0 + j];
            afrag[kk][j + 8] = (_Float16)arow[k1 + j];
        }
    }

    v8f acc[8];
    #pragma unroll
    for (int n = 0; n < 8; ++n) {
        v8f c = {};
        const int ncol = n * 16 + (lane & 15);
        #pragma unroll
        for (int kk = 0; kk < 4; ++kk) {
            // B fragment: lane holds column ncol; 16 contiguous K values from Wt
            const v16h b = *(const v16h*)&Wt[ncol * C_DIM + kk * 32 + half * 16];
            c = __builtin_amdgcn_wmma_f32_16x16x32_f16(
                    false, afrag[kk], false, b, (short)0, c, false, false);
        }
        acc[n] = c;
    }

    // C/D layout: VGPR r holds M = r + 8*half, N = n*16 + lane%16
    float* drow = dst + (size_t)mtile * 16 * C_DIM;
    #pragma unroll
    for (int n = 0; n < 8; ++n) {
        #pragma unroll
        for (int r = 0; r < 8; ++r) {
            drow[(size_t)(r + 8 * half) * C_DIM + n * 16 + (lane & 15)] = acc[n][r];
        }
    }
}

// ---------------------------------------------------------------- propagation
// out[i] = dis[i]^2 * h[i]   (self-loop term; also initializes out)
__global__ void selfloop_kernel(const float* __restrict__ h,
                                const float* __restrict__ dis,
                                float* __restrict__ out, int N) {
    int t = blockIdx.x * blockDim.x + threadIdx.x;
    int i = t >> 5, lane = t & 31;
    if (i >= N) return;
    float s = dis[i]; s = s * s;
    float4 v = ((const float4*)(h + (size_t)i * C_DIM))[lane];
    float4 o; o.x = s * v.x; o.y = s * v.y; o.z = s * v.z; o.w = s * v.w;
    ((float4*)(out + (size_t)i * C_DIM))[lane] = o;
}

// out[col[e]] += dis[row]*w*dis[col] * h[row[e]] ; one wave per edge,
// float4 gather per lane (128 floats/row), f32 global atomics (L2-resident).
__global__ void propagate_kernel(const float* __restrict__ h,
                                 float* __restrict__ out,
                                 const int* __restrict__ row,
                                 const int* __restrict__ col,
                                 const float* __restrict__ w,
                                 const float* __restrict__ dis, int E) {
    int wid  = (blockIdx.x * blockDim.x + threadIdx.x) >> 5;
    int lane = threadIdx.x & 31;
    if (wid >= E) return;
    int r = row[wid], c = col[wid];
    float nrm = dis[r] * w[wid] * dis[c];
    float4 v = ((const float4*)(h + (size_t)r * C_DIM))[lane];
    float* oc = out + (size_t)c * C_DIM + lane * 4;
    atomicAdd(oc + 0, nrm * v.x);
    atomicAdd(oc + 1, nrm * v.y);
    atomicAdd(oc + 2, nrm * v.z);
    atomicAdd(oc + 3, nrm * v.w);
}

// ---------------------------------------------------------------- bias + BN + ReLU
__global__ void bn_relu_kernel(float* __restrict__ x,
                               const float* __restrict__ b,
                               const float* __restrict__ g,
                               const float* __restrict__ be,
                               const float* __restrict__ rm,
                               const float* __restrict__ rv,
                               long long total) {
    long long i = (long long)blockIdx.x * blockDim.x + threadIdx.x;
    if (i >= total) return;
    int c = (int)(i & (C_DIM - 1));
    float v = x[i] + b[c];
    v = (v - rm[c]) * rsqrtf(rv[c] + 1e-5f) * g[c] + be[c];
    x[i] = v > 0.0f ? v : 0.0f;
}

// ---------------------------------------------------------------- pooling + linear
__global__ void zero_pool_kernel(float* __restrict__ pooled, float* __restrict__ cnt) {
    int i = blockIdx.x * blockDim.x + threadIdx.x;
    if (i < NGRAPH * C_DIM) pooled[i] = 0.0f;
    if (i < NGRAPH) cnt[i] = 0.0f;
}

__global__ void pool_accum_kernel(const float* __restrict__ h,
                                  const int* __restrict__ batch,
                                  float* __restrict__ pooled,
                                  float* __restrict__ cnt, int N) {
    int t = blockIdx.x * blockDim.x + threadIdx.x;
    int i = t >> 5, lane = t & 31;
    if (i >= N) return;
    int b = batch[i];
    float4 v = ((const float4*)(h + (size_t)i * C_DIM))[lane];
    float* p = pooled + (size_t)b * C_DIM + lane * 4;
    atomicAdd(p + 0, v.x); atomicAdd(p + 1, v.y);
    atomicAdd(p + 2, v.z); atomicAdd(p + 3, v.w);
    if (lane == 0) atomicAdd(&cnt[b], 1.0f);
}

__global__ void final_linear_kernel(const float* __restrict__ pooled,
                                    const float* __restrict__ cnt,
                                    const float* __restrict__ Wl,
                                    const float* __restrict__ bl,
                                    float* __restrict__ out) {
    int t = blockIdx.x * blockDim.x + threadIdx.x;  // 512*2 threads
    if (t >= NGRAPH * 2) return;
    int g = t >> 1, o = t & 1;
    float inv = 1.0f / fmaxf(cnt[g], 1.0f);
    float s = bl[o];
    const float* pg = pooled + (size_t)g * C_DIM;
    #pragma unroll 8
    for (int k = 0; k < C_DIM; ++k) s += (pg[k] * inv) * Wl[k * 2 + o];
    out[t] = s;
}

// ---------------------------------------------------------------- launcher
extern "C" void kernel_launch(void* const* d_in, const int* in_sizes, int n_in,
                              void* d_out, int out_size, void* d_ws, size_t ws_size,
                              hipStream_t stream) {
    const float* x     = (const float*)d_in[0];
    const int*   eidx  = (const int*)d_in[1];
    const float* ew    = (const float*)d_in[2];
    const int*   batch = (const int*)d_in[3];
    const float* W[3]  = {(const float*)d_in[4],  (const float*)d_in[10], (const float*)d_in[16]};
    const float* b[3]  = {(const float*)d_in[5],  (const float*)d_in[11], (const float*)d_in[17]};
    const float* g[3]  = {(const float*)d_in[6],  (const float*)d_in[12], (const float*)d_in[18]};
    const float* be[3] = {(const float*)d_in[7],  (const float*)d_in[13], (const float*)d_in[19]};
    const float* rm[3] = {(const float*)d_in[8],  (const float*)d_in[14], (const float*)d_in[20]};
    const float* rv[3] = {(const float*)d_in[9],  (const float*)d_in[15], (const float*)d_in[21]};
    const float* Wl    = (const float*)d_in[22];
    const float* bl    = (const float*)d_in[23];

    const int N = in_sizes[0] / C_DIM;        // 100000 (divisible by 16)
    const int E = in_sizes[1] / 2;            // 1600000
    const int* row = eidx;
    const int* col = eidx + E;

    // workspace partition (floats): deg | dis | pooled | cnt | bufA | bufB
    float* deg    = (float*)d_ws;
    float* dis    = deg + N;
    float* pooled = dis + N;
    float* cnt    = pooled + (size_t)NGRAPH * C_DIM;
    float* bufA   = cnt + NGRAPH;                       // GEMM output h
    float* bufB   = bufA + (size_t)N * C_DIM;           // propagated / activated

    const int T = 256;

    // GCN normalization
    init_deg_kernel <<<(N + T - 1) / T, T, 0, stream>>>(deg, N);
    accum_deg_kernel<<<(E + T - 1) / T, T, 0, stream>>>(col, ew, deg, E);
    dis_kernel      <<<(N + T - 1) / T, T, 0, stream>>>(deg, dis, N);
    zero_pool_kernel<<<(NGRAPH * C_DIM + T - 1) / T, T, 0, stream>>>(pooled, cnt);

    const long long totalNC = (long long)N * C_DIM;
    const int gemm_blocks  = ((N >> 4) + 7) / 8;
    const int node_wave_bl = (N * 32 + T - 1) / T;
    const int edge_wave_bl = (int)(((long long)E * 32 + T - 1) / T);
    const int nc_blocks    = (int)((totalNC + T - 1) / T);

    const float* src = x;
    for (int l = 0; l < 3; ++l) {
        gemm_wmma_kernel<<<gemm_blocks, T, 0, stream>>>(src, W[l], bufA, N);
        selfloop_kernel <<<node_wave_bl, T, 0, stream>>>(bufA, dis, bufB, N);
        propagate_kernel<<<edge_wave_bl, T, 0, stream>>>(bufA, bufB, row, col, ew, dis, E);
        bn_relu_kernel  <<<nc_blocks, T, 0, stream>>>(bufB, b[l], g[l], be[l], rm[l], rv[l], totalNC);
        src = bufB;
    }

    pool_accum_kernel  <<<node_wave_bl, T, 0, stream>>>(bufB, batch, pooled, cnt, N);
    final_linear_kernel<<<(NGRAPH * 2 + T - 1) / T, T, 0, stream>>>(pooled, cnt, Wl, bl, (float*)d_out);
}
